// MoE_59270548685030
// MI455X (gfx1250) — matrix operand order repaired and checked by reading, not compiled, not run on hardware
//
#include <hip/hip_runtime.h>
#include <hip/hip_bf16.h>

#define STOK 8192
#define HDIM 1024
#define NEXP 8
#define DFFD 4096
#define CAP  1024

typedef __attribute__((ext_vector_type(16))) __bf16 v16bf;
typedef __attribute__((ext_vector_type(8)))  float  v8f;

// CDNA5 async global->LDS copy (ASYNCcnt-tracked, no VGPR staging).
// LDS address = low 32 bits of generic pointer (ISA 10.2).
#define ASYNC_LOAD_B128(lds_off, gaddr64)                                  \
    asm volatile("global_load_async_to_lds_b128 %0, %1, off"               \
                 :: "v"(lds_off), "v"(gaddr64) : "memory")
#define WAIT_ASYNCCNT(n)                                                   \
    asm volatile("s_wait_asynccnt " #n ::: "memory")

__device__ __forceinline__ unsigned short f2bf(float f) {
    unsigned int u = __float_as_uint(f);
    u += 0x7fffu + ((u >> 16) & 1u);     // round-to-nearest-even
    return (unsigned short)(u >> 16);
}

// ---------------------------------------------------------------- zero init
__global__ void moe_zero_kernel(float* p, int n) {
    for (int i = blockIdx.x * blockDim.x + threadIdx.x; i < n; i += gridDim.x * blockDim.x)
        p[i] = 0.0f;
}

// ---------------------------------------------------------------- gating
// one wave (32 lanes) per token: logits = x[s,:] @ wg[:,0..7], softmax, argmax
__global__ __launch_bounds__(256)
void moe_gating_kernel(const float* __restrict__ x, const float* __restrict__ wg,
                       float* __restrict__ gates_all, float* __restrict__ gate_val,
                       int* __restrict__ expert_idx) {
    const int wave = threadIdx.x >> 5;
    const int lane = threadIdx.x & 31;
    const int s = blockIdx.x * 8 + wave;
    const float* xr = x + (size_t)s * HDIM;

    float acc[8];
#pragma unroll
    for (int e = 0; e < 8; ++e) acc[e] = 0.0f;

    for (int h = lane; h < HDIM; h += 32) {
        float xv = xr[h];
        const float4* wr = (const float4*)(wg + (size_t)h * NEXP);
        float4 g0 = wr[0];
        float4 g1 = wr[1];
        acc[0] += xv * g0.x; acc[1] += xv * g0.y; acc[2] += xv * g0.z; acc[3] += xv * g0.w;
        acc[4] += xv * g1.x; acc[5] += xv * g1.y; acc[6] += xv * g1.z; acc[7] += xv * g1.w;
    }
#pragma unroll
    for (int off = 16; off > 0; off >>= 1)
#pragma unroll
        for (int e = 0; e < 8; ++e) acc[e] += __shfl_xor(acc[e], off, 32);

    if (lane == 0) {
        float mx = acc[0];
#pragma unroll
        for (int e = 1; e < 8; ++e) mx = fmaxf(mx, acc[e]);
        float p[8], den = 0.0f;
#pragma unroll
        for (int e = 0; e < 8; ++e) { p[e] = __expf(acc[e] - mx); den += p[e]; }
        float inv = 1.0f / den;
        int best = 0; float bg = -1.0f;
#pragma unroll
        for (int e = 0; e < 8; ++e) {
            p[e] *= inv;
            gates_all[(size_t)s * 8 + e] = p[e];
            if (p[e] > bg) { bg = p[e]; best = e; }
        }
        gate_val[s] = bg;
        expert_idx[s] = best;
    }
}

// ---------------------------------------------------------------- capacity scan
// one block per expert; deterministic token-order prefix sum (Hillis-Steele in LDS)
__global__ __launch_bounds__(1024)
void moe_scan_kernel(const int* __restrict__ expert_idx, int* __restrict__ slot_token,
                     int* __restrict__ counts_pre) {
    __shared__ int sh[1024];
    const int e = blockIdx.x;
    const int tid = threadIdx.x;
    int base = 0;
    for (int c0 = 0; c0 < STOK; c0 += 1024) {
        int s = c0 + tid;
        int flag = (expert_idx[s] == e) ? 1 : 0;
        sh[tid] = flag;
        __syncthreads();
        for (int off = 1; off < 1024; off <<= 1) {
            int t = (tid >= off) ? sh[tid - off] : 0;
            __syncthreads();
            sh[tid] += t;
            __syncthreads();
        }
        int pos = base + sh[tid] - flag;   // exclusive prefix
        if (flag && pos < CAP) slot_token[e * CAP + pos] = s;
        base += sh[1023];
        __syncthreads();
    }
    if (tid == 0) counts_pre[e] = base;
    int cnt = base < CAP ? base : CAP;
    for (int p = cnt + tid; p < CAP; p += 1024) slot_token[e * CAP + p] = -1;
}

// ---------------------------------------------------------------- l_aux + exp_counts
__global__ __launch_bounds__(256)
void moe_finalize_kernel(const float* __restrict__ gates_all, const int* __restrict__ counts_pre,
                         float* __restrict__ d_tail) {
    __shared__ float red[8][256];
    const int tid = threadIdx.x;
    float a[8];
#pragma unroll
    for (int e = 0; e < 8; ++e) a[e] = 0.0f;
    for (int s = tid; s < STOK; s += 256) {
        const float* g = gates_all + (size_t)s * 8;
#pragma unroll
        for (int e = 0; e < 8; ++e) a[e] += g[e];
    }
#pragma unroll
    for (int e = 0; e < 8; ++e) red[e][tid] = a[e];
    __syncthreads();
    for (int off = 128; off > 0; off >>= 1) {
        if (tid < off)
#pragma unroll
            for (int e = 0; e < 8; ++e) red[e][tid] += red[e][tid + off];
        __syncthreads();
    }
    if (tid == 0) {
        float l = 0.0f;
#pragma unroll
        for (int e = 0; e < 8; ++e) l += red[e][0] * (float)counts_pre[e];
        d_tail[0] = l * (float)NEXP / ((float)STOK * (float)STOK);
    }
    if (tid < NEXP) {
        int c = counts_pre[tid];
        if (c > CAP) c = CAP;
        d_tail[1 + tid] = (float)c;
    }
}

// ---------------------------------------------------------------- gather + f32->bf16
// one block per (expert, slot); empty slots -> zero rows
__global__ __launch_bounds__(256)
void moe_gather_kernel(const float* __restrict__ x, const int* __restrict__ slot_token,
                       unsigned short* __restrict__ Xe) {
    const int slot = blockIdx.x;            // e*CAP + c
    const int t = slot_token[slot];
    const int i = threadIdx.x * 4;
    unsigned short r0 = 0, r1 = 0, r2 = 0, r3 = 0;
    if (t >= 0) {
        float4 v = *(const float4*)(x + (size_t)t * HDIM + i);
        r0 = f2bf(v.x); r1 = f2bf(v.y); r2 = f2bf(v.z); r3 = f2bf(v.w);
    }
    uint2 o;
    o.x = (unsigned)r0 | ((unsigned)r1 << 16);
    o.y = (unsigned)r2 | ((unsigned)r3 << 16);
    *(uint2*)(Xe + (size_t)slot * HDIM + i) = o;
}

// ---------------------------------------------------------------- weight transpose+convert
// w [E][K][N] f32 -> wt [E][N][K] bf16, LDS-tiled, coalesced both directions
__global__ __launch_bounds__(256)
void moe_transpose_kernel(const float* __restrict__ w, unsigned short* __restrict__ wt,
                          int K, int N) {
    __shared__ float tile[32][33];
    const int e = blockIdx.z;
    const int n0 = blockIdx.x * 32, k0 = blockIdx.y * 32;
    const int tx = threadIdx.x, ty = threadIdx.y;      // (32,8)
    const float* we = w + (size_t)e * K * N;
    unsigned short* wte = wt + (size_t)e * (size_t)N * K;
#pragma unroll
    for (int i = 0; i < 4; ++i)
        tile[ty + i * 8][tx] = we[(size_t)(k0 + ty + i * 8) * N + n0 + tx];
    __syncthreads();
#pragma unroll
    for (int i = 0; i < 4; ++i)
        wte[(size_t)(n0 + ty + i * 8) * K + k0 + tx] = f2bf(tile[tx][ty + i * 8]);
}

// ---------------------------------------------------------------- WMMA GEMM
// D[e] = A[e] (MxK bf16 row-major) * B[e]^T (B stored [N][K] bf16) + bias
// FUSE_GELU=1: GELU epilogue, bf16 store to Obf[e][M][N]
// FUSE_GELU=0: gate-weighted scatter to out[token][HDIM]
// Block tile 128(M) x 128(N), BK=32; 8 waves as 2x4, each wave 64x32 (4x2 wmma frags).
// Double-buffered LDS fed by global_load_async_to_lds_b128 (ASYNCcnt pipeline).
template <int FUSE_GELU>
__global__ __launch_bounds__(256)
void moe_gemm_kernel(const unsigned short* __restrict__ A,
                     const unsigned short* __restrict__ B,
                     const float* __restrict__ bias,
                     unsigned short* __restrict__ Obf,
                     float* __restrict__ out,
                     const int* __restrict__ slot_token,
                     const float* __restrict__ gate_val,
                     int M, int N, int K) {
    // padded row stride 40 halfwords (80B) -> conflict-free b128 fragment reads
    __shared__ unsigned short As[2 * 128 * 40];
    __shared__ unsigned short Bs[2 * 128 * 40];
    const unsigned ABUF = 128 * 40 * 2;   // bytes per buffer

    const int e = blockIdx.z;
    const int n0 = blockIdx.x * 128;
    const int m0 = blockIdx.y * 128;
    const unsigned short* Ae = A + (size_t)e * (size_t)M * K;
    const unsigned short* Be = B + (size_t)e * (size_t)N * K;

    const int tid = threadIdx.x;
    const int lane = tid & 31;
    const int wave = tid >> 5;
    const int wm = wave >> 2, wn = wave & 3;   // 2 x 4 wave grid
    const int lr = lane & 15, sel = lane >> 4;

    // cooperative loader: 128 rows x 32 cols per matrix, 2 x b128 per thread each
    const int lrow = tid >> 1;             // 0..127
    const int lcol = (tid & 1) * 16;       // 0 / 16

    const unsigned asBase = (unsigned)(size_t)(&As[0]);
    const unsigned bsBase = (unsigned)(size_t)(&Bs[0]);

    v8f acc[4][2];
#pragma unroll
    for (int fm = 0; fm < 4; ++fm)
#pragma unroll
        for (int fn = 0; fn < 2; ++fn)
#pragma unroll
            for (int j = 0; j < 8; ++j) acc[fm][fn][j] = 0.0f;

    union Frag { uint4 u[2]; v16bf v; };

    auto issue_tile = [&](int k0, int buf) {
        const unsigned short* ag = Ae + (size_t)(m0 + lrow) * K + k0 + lcol;
        const unsigned short* bg = Be + (size_t)(n0 + lrow) * K + k0 + lcol;
        unsigned aoff = asBase + (unsigned)buf * ABUF + (unsigned)(lrow * 40 + lcol) * 2;
        unsigned boff = bsBase + (unsigned)buf * ABUF + (unsigned)(lrow * 40 + lcol) * 2;
        unsigned long long ga0 = (unsigned long long)(size_t)ag;
        unsigned long long ga1 = (unsigned long long)(size_t)(ag + 8);
        unsigned long long gb0 = (unsigned long long)(size_t)bg;
        unsigned long long gb1 = (unsigned long long)(size_t)(bg + 8);
        ASYNC_LOAD_B128(aoff, ga0);
        ASYNC_LOAD_B128(aoff + 16, ga1);
        ASYNC_LOAD_B128(boff, gb0);
        ASYNC_LOAD_B128(boff + 16, gb1);
    };

    issue_tile(0, 0);
    const int nk = K / 32;
    for (int i = 0; i < nk; ++i) {
        const int buf = i & 1;
        if (i + 1 < nk) {
            issue_tile((i + 1) * 32, buf ^ 1);   // prefetch next slice into other buffer
            WAIT_ASYNCCNT(4);                    // retire current slice (in-order completion)
        } else {
            WAIT_ASYNCCNT(0);
        }
        __syncthreads();

        const unsigned short* ab = &As[buf * 128 * 40];
        const unsigned short* bb = &Bs[buf * 128 * 40];

        Frag af[4], bfm[2];
#pragma unroll
        for (int fm = 0; fm < 4; ++fm) {
            // 16-bit A layout: lanes 0-15 K=0..7,16..23 ; lanes 16-31 K=8..15,24..31
            const unsigned short* p = &ab[(wm * 64 + fm * 16 + lr) * 40 + sel * 8];
            af[fm].u[0] = *(const uint4*)p;
            af[fm].u[1] = *(const uint4*)(p + 16);
        }
#pragma unroll
        for (int fn = 0; fn < 2; ++fn) {
            // B layout: lane n holds column n; lanes 0-15 K=0..7,16..23 ; lanes 16-31 K=8..15,24..31
            const unsigned short* p = &bb[(wn * 32 + fn * 16 + lr) * 40 + sel * 8];
            bfm[fn].u[0] = *(const uint4*)p;
            bfm[fn].u[1] = *(const uint4*)(p + 16);
        }
#pragma unroll
        for (int fm = 0; fm < 4; ++fm)
#pragma unroll
            for (int fn = 0; fn < 2; ++fn)
                acc[fm][fn] = __builtin_amdgcn_wmma_f32_16x16x32_bf16(
                    false, af[fm].v, false, bfm[fn].v, (short)0, acc[fm][fn], false, false);
        __syncthreads();   // safe to overwrite this buffer next iteration
    }

    // epilogue: C/D layout -> VGPR j, lanes 0-15: M=j, lanes 16-31: M=j+8; N=lane&15
#pragma unroll
    for (int fm = 0; fm < 4; ++fm) {
#pragma unroll
        for (int j = 0; j < 8; ++j) {
            const int row = m0 + wm * 64 + fm * 16 + sel * 8 + j;
            int tok = -1; float g = 0.0f;
            if (!FUSE_GELU) {
                tok = slot_token[e * M + row];
                if (tok >= 0) g = gate_val[tok];
            }
#pragma unroll
            for (int fn = 0; fn < 2; ++fn) {
                const int col = n0 + wn * 32 + fn * 16 + lr;
                float v = acc[fm][fn][j] + bias[(size_t)e * N + col];
                if (FUSE_GELU) {
                    // tanh-approx GELU (jax.nn.gelu default)
                    float c = v + 0.044715f * v * v * v;
                    float gl = 0.5f * v * (1.0f + tanhf(0.7978845608028654f * c));
                    Obf[(size_t)e * (size_t)M * N + (size_t)row * N + col] = f2bf(gl);
                } else {
                    if (tok >= 0) out[(size_t)tok * HDIM + col] = g * v;
                }
            }
        }
    }
}

// ----------------------------------------------------------------
extern "C" void kernel_launch(void* const* d_in, const int* in_sizes, int n_in,
                              void* d_out, int out_size, void* d_ws, size_t ws_size,
                              hipStream_t stream) {
    const float* x  = (const float*)d_in[0];   // [S,H]
    const float* wg = (const float*)d_in[1];   // [H,E]
    const float* w1 = (const float*)d_in[2];   // [E,H,DFF]
    const float* b1 = (const float*)d_in[3];   // [E,DFF]
    const float* w2 = (const float*)d_in[4];   // [E,DFF,H]
    const float* b2 = (const float*)d_in[5];   // [E,H]
    float* out = (float*)d_out;                // [S*H] out, [1] l_aux, [E] exp_counts

    char* ws = (char*)d_ws;
    size_t off = 0;
    auto alloc = [&](size_t bytes) -> void* {
        void* p = ws + off;
        off = (off + bytes + 255) & ~(size_t)255;
        return p;
    };
    float* gates_all      = (float*)alloc((size_t)STOK * NEXP * 4);
    float* gate_val       = (float*)alloc((size_t)STOK * 4);
    int*   expert_idx     = (int*)  alloc((size_t)STOK * 4);
    int*   slot_token     = (int*)  alloc((size_t)NEXP * CAP * 4);
    int*   counts_pre     = (int*)  alloc((size_t)NEXP * 4);
    unsigned short* Xe    = (unsigned short*)alloc((size_t)NEXP * CAP * HDIM * 2);
    unsigned short* W1t   = (unsigned short*)alloc((size_t)NEXP * DFFD * HDIM * 2);  // [E][DFF][H]
    unsigned short* W2t   = (unsigned short*)alloc((size_t)NEXP * HDIM * DFFD * 2);  // [E][H][DFF]
    unsigned short* H1    = (unsigned short*)alloc((size_t)NEXP * CAP * DFFD * 2);   // [E][C][DFF]

    // 1. zero the token-major output (dropped tokens must be 0)
    moe_zero_kernel<<<4096, 256, 0, stream>>>(out, STOK * HDIM + 1 + NEXP);
    // 2. gating
    moe_gating_kernel<<<STOK / 8, 256, 0, stream>>>(x, wg, gates_all, gate_val, expert_idx);
    // 3. deterministic capacity assignment
    moe_scan_kernel<<<NEXP, 1024, 0, stream>>>(expert_idx, slot_token, counts_pre);
    // 4. l_aux + exp_counts
    moe_finalize_kernel<<<1, 256, 0, stream>>>(gates_all, counts_pre, out + (size_t)STOK * HDIM);
    // 5. gather dispatched tokens to bf16
    moe_gather_kernel<<<NEXP * CAP, 256, 0, stream>>>(x, slot_token, Xe);
    // 6. weights -> bf16, transposed to [N][K]
    {
        dim3 b(32, 8);
        dim3 g1(DFFD / 32, HDIM / 32, NEXP);  // w1: K=H, N=DFF
        moe_transpose_kernel<<<g1, b, 0, stream>>>(w1, W1t, HDIM, DFFD);
        dim3 g2(HDIM / 32, DFFD / 32, NEXP);  // w2: K=DFF, N=H
        moe_transpose_kernel<<<g2, b, 0, stream>>>(w2, W2t, DFFD, HDIM);
    }
    // 7. GEMM1 + bias + GELU -> H1 (bf16)
    {
        dim3 g(DFFD / 128, CAP / 128, NEXP);
        moe_gemm_kernel<1><<<g, 256, 0, stream>>>(Xe, W1t, b1, H1, nullptr, nullptr, nullptr,
                                                  CAP, DFFD, HDIM);
    }
    // 8. GEMM2 + bias + gate-weighted scatter -> out
    {
        dim3 g(HDIM / 128, CAP / 128, NEXP);
        moe_gemm_kernel<0><<<g, 256, 0, stream>>>(H1, W2t, b2, nullptr, out, slot_token, gate_val,
                                                  CAP, HDIM, DFFD);
    }
    (void)in_sizes; (void)n_in; (void)out_size; (void)ws_size;
}
